// PhraseEncoder_65335042507168
// MI455X (gfx1250) — matrix operand order repaired and checked by reading, not compiled, not run on hardware
//
#include <hip/hip_runtime.h>
#include <math.h>

typedef __attribute__((ext_vector_type(16))) _Float16 v16h;
typedef __attribute__((ext_vector_type(8)))  float    v8f;

union FragU { uint4 q[2]; v16h v; };
union Pack8 { uint4 q; _Float16 h[8]; };
union Pack4 { uint2 u; _Float16 h[4]; };

__device__ __forceinline__ float gelu_erf(float x) {
    return 0.5f * x * (1.0f + erff(x * 0.70710678118654752f));
}

struct Params {
    const int*   comp_ids;
    const int*   change_ids;
    const float* bbox;
    const float* comp_table;
    const float* change_table;
    const float* pos_w1; const float* pos_b1; const float* pos_g1; const float* pos_be1;
    const float* pos_w2; const float* pos_b2;
    const float* fus_b1;  const float* fus_g1;  const float* fus_be1;
    const float* fus_b2;  const float* fus_g2;  const float* fus_be2;
    const float* proj_b1; const float* proj_g1; const float* proj_be1;
    const float* proj_b2; const float* proj_g2; const float* proj_be2;
    const _Float16* wt_fus1;   // [256][160]
    const _Float16* wt_fus2;   // [768][256]
    const _Float16* wt_proj1;  // [512][768]
    const _Float16* wt_proj2;  // [768][512]
    float* out;
};

// Convert fp32 [K][Nout] row-major -> f16 [Nout][K] (transposed), so each
// WMMA B-fragment (one output column, contiguous K) is two 16B loads.
__global__ void transpose_f32_to_f16(const float* __restrict__ src,
                                     _Float16* __restrict__ dst,
                                     int K, int Nout) {
    int idx = blockIdx.x * blockDim.x + threadIdx.x;
    if (idx >= K * Nout) return;
    int n = idx / K;
    int k = idx - n * K;
    dst[n * K + k] = (_Float16)src[k * Nout + n];
}

// One layer: D = LN(gelu?)(A[16,Kd] @ W[Kd,Nd] + b).  A in LDS f16, W from
// global f16 transposed [Nd][Kd].  Non-final: f16 result -> bufOut (LDS).
// Final: f32 pre-LN -> gOut (global), LN applied in place.
template<int Kd, int Nd, bool DOGELU, bool FINAL>
__device__ __forceinline__ void mlp_layer(
    const _Float16* bufIn, _Float16* bufOut, float* gOut,
    const _Float16* __restrict__ Wt,
    const float* __restrict__ bias,
    const float* __restrict__ gamma,
    const float* __restrict__ beta,
    float* sMean, float* sRstd, int lane)
{
    const int half = lane >> 4;     // which 16-lane half
    const int m    = lane & 15;     // column-in-tile / row-in-tile selector
    float sum[8], sumsq[8];
#pragma unroll
    for (int j = 0; j < 8; ++j) { sum[j] = 0.f; sumsq[j] = 0.f; }

    const _Float16* ap0 = bufIn + m * Kd + half * 8;   // A: row m, K-halves interleaved by 8

#pragma unroll 1
    for (int nt = 0; nt < Nd / 16; ++nt) {
        const int n = nt * 16 + m;                      // output column for this lane
        const float bv = bias[n];
        v8f acc;
#pragma unroll
        for (int j = 0; j < 8; ++j) acc[j] = bv;        // C starts at bias (LN sees x@W+b)

        const _Float16* wp = Wt + (size_t)n * Kd + half * 8;   // B: col n, K-halves interleaved by 8
#pragma unroll 4
        for (int kt = 0; kt < Kd / 32; ++kt) {
            FragU a, b;
            const _Float16* ap = ap0 + kt * 32;
            a.q[0] = *reinterpret_cast<const uint4*>(ap);        // K = k0+half*8 .. +7
            a.q[1] = *reinterpret_cast<const uint4*>(ap + 16);   // K = k0+16+half*8 .. +7
            const _Float16* bp = wp + kt * 32;
            b.q[0] = *reinterpret_cast<const uint4*>(bp);        // K = k0+half*8 .. +7
            b.q[1] = *reinterpret_cast<const uint4*>(bp + 16);   // K = k0+16+half*8 .. +7
            acc = __builtin_amdgcn_wmma_f32_16x16x32_f16(
                false, a.v, false, b.v, (short)0, acc, false, false);
        }
        // D layout: VGPR j -> row (half*8+j), column n.  Accumulate LN stats.
#pragma unroll
        for (int j = 0; j < 8; ++j) {
            const float d = acc[j];
            sum[j]   += d;
            sumsq[j] += d * d;
            const int rr = half * 8 + j;
            if (FINAL) gOut[rr * Nd + n] = d;
            else       bufOut[rr * Nd + n] = (_Float16)d;
        }
    }

    // Row stats: reduce across the 16 lanes of each half (rows differ by half).
#pragma unroll
    for (int mask = 1; mask <= 8; mask <<= 1) {
#pragma unroll
        for (int j = 0; j < 8; ++j) {
            sum[j]   += __shfl_xor(sum[j],   mask, 32);
            sumsq[j] += __shfl_xor(sumsq[j], mask, 32);
        }
    }
    if (m == 0) {
#pragma unroll
        for (int j = 0; j < 8; ++j) {
            const int rr = half * 8 + j;
            const float mean = sum[j] * (1.0f / Nd);
            float var = sumsq[j] * (1.0f / Nd) - mean * mean;
            var = fmaxf(var, 0.0f);
            sMean[rr] = mean;
            sRstd[rr] = rsqrtf(var + 1e-5f);
        }
    }
    asm volatile("s_wait_dscnt 0" ::: "memory");   // same-wave LDS visibility

    // Pass 2: normalize (+ gelu), 8 contiguous f16/f32 per lane per step.
    // 32 lanes * 8 = 256-wide sweep; Nd is a multiple of 256.
#pragma unroll 1
    for (int c0 = lane * 8; c0 < Nd; c0 += 256) {
        float ga[8], be[8];
        *(float4*)&ga[0] = *(const float4*)(gamma + c0);
        *(float4*)&ga[4] = *(const float4*)(gamma + c0 + 4);
        *(float4*)&be[0] = *(const float4*)(beta + c0);
        *(float4*)&be[4] = *(const float4*)(beta + c0 + 4);
#pragma unroll 1
        for (int r = 0; r < 16; ++r) {
            const float mean = sMean[r];
            const float rstd = sRstd[r];
            float v[8];
            if (FINAL) {
                float* pp = gOut + r * Nd + c0;
                *(float4*)&v[0] = *(const float4*)pp;
                *(float4*)&v[4] = *(const float4*)(pp + 4);
#pragma unroll
                for (int t = 0; t < 8; ++t) {
                    v[t] = (v[t] - mean) * rstd * ga[t] + be[t];
                    if (DOGELU) v[t] = gelu_erf(v[t]);
                }
                *(float4*)pp       = *(float4*)&v[0];
                *(float4*)(pp + 4) = *(float4*)&v[4];
            } else {
                Pack8 pk;
                pk.q = *reinterpret_cast<const uint4*>(&bufOut[r * Nd + c0]);
#pragma unroll
                for (int t = 0; t < 8; ++t) {
                    float x = (float)pk.h[t];
                    x = (x - mean) * rstd * ga[t] + be[t];
                    if (DOGELU) x = gelu_erf(x);
                    pk.h[t] = (_Float16)x;
                }
                *reinterpret_cast<uint4*>(&bufOut[r * Nd + c0]) = pk.q;
            }
        }
    }
    asm volatile("s_wait_dscnt 0" ::: "memory");
}

__global__ void __launch_bounds__(32)
phrase_encoder_kernel(Params p)
{
    __shared__ _Float16 bufA[16 * 768];   // x160 / h768 / (dummy)
    __shared__ _Float16 bufB[16 * 512];   // h256 / y512
    __shared__ float sMean[16];
    __shared__ float sRstd[16];

    const int lane = threadIdx.x;
    const int row0 = blockIdx.x * 16;

    // ---- Stage 0a: pos-MLP (4 -> 32 -> LN -> gelu -> 64), 2 lanes per row.
    {
        const int r   = lane >> 1;       // row 0..15
        const int sub = lane & 1;        // which half of the 64 outputs
        const int rg  = row0 + r;
        float b[4];
#pragma unroll
        for (int i = 0; i < 4; ++i)
            b[i] = fminf(fmaxf(p.bbox[rg * 4 + i], 0.0f), 1.0f);
        float h[32];
        float s = 0.f, s2 = 0.f;
#pragma unroll
        for (int j = 0; j < 32; ++j) {
            float v = p.pos_b1[j];
#pragma unroll
            for (int i = 0; i < 4; ++i) v += b[i] * p.pos_w1[i * 32 + j];
            h[j] = v; s += v; s2 += v * v;
        }
        const float mean = s * (1.0f / 32.0f);
        float var = s2 * (1.0f / 32.0f) - mean * mean;
        var = fmaxf(var, 0.0f);
        const float rstd = rsqrtf(var + 1e-5f);
#pragma unroll
        for (int j = 0; j < 32; ++j)
            h[j] = gelu_erf((h[j] - mean) * rstd * p.pos_g1[j] + p.pos_be1[j]);
        // this lane produces outputs kk = sub*32 .. sub*32+31, packed 4 at a time
#pragma unroll 1
        for (int k4 = 0; k4 < 8; ++k4) {
            Pack4 pk;
#pragma unroll
            for (int t = 0; t < 4; ++t) {
                const int kk = sub * 32 + k4 * 4 + t;
                float v = p.pos_b2[kk];
#pragma unroll
                for (int j = 0; j < 32; ++j) v += h[j] * p.pos_w2[j * 64 + kk];
                pk.h[t] = (_Float16)v;
            }
            *reinterpret_cast<uint2*>(&bufA[r * 160 + 96 + sub * 32 + k4 * 4]) = pk.u;
        }
    }

    // ---- Stage 0b: embedding gathers, whole wave, float4 chunks -> packed f16.
    // 16 rows * 24 float4-chunks (16 comp + 8 change) = 384 chunks.
#pragma unroll 1
    for (int e = lane; e < 384; e += 32) {
        const int r  = e / 24;
        const int ch = e - r * 24;
        const int rg = row0 + r;
        const float* src;
        int dstoff;
        if (ch < 16) {
            src = p.comp_table + p.comp_ids[rg] * 64 + ch * 4;
            dstoff = r * 160 + ch * 4;
        } else {
            src = p.change_table + p.change_ids[rg] * 32 + (ch - 16) * 4;
            dstoff = r * 160 + 64 + (ch - 16) * 4;
        }
        const float4 v = *reinterpret_cast<const float4*>(src);
        Pack4 pk;
        pk.h[0] = (_Float16)v.x; pk.h[1] = (_Float16)v.y;
        pk.h[2] = (_Float16)v.z; pk.h[3] = (_Float16)v.w;
        *reinterpret_cast<uint2*>(&bufA[dstoff]) = pk.u;
    }
    asm volatile("s_wait_dscnt 0" ::: "memory");

    float* gOut = p.out + (size_t)row0 * 768;

    // ---- 4 fused GEMM+LN(+GELU) layers, all WMMA f16->f32
    mlp_layer<160, 256, true,  false>(bufA, bufB, gOut, p.wt_fus1,
        p.fus_b1, p.fus_g1, p.fus_be1, sMean, sRstd, lane);
    mlp_layer<256, 768, true,  false>(bufB, bufA, gOut, p.wt_fus2,
        p.fus_b2, p.fus_g2, p.fus_be2, sMean, sRstd, lane);
    mlp_layer<768, 512, true,  false>(bufA, bufB, gOut, p.wt_proj1,
        p.proj_b1, p.proj_g1, p.proj_be1, sMean, sRstd, lane);
    mlp_layer<512, 768, false, true >(bufB, bufA, gOut, p.wt_proj2,
        p.proj_b2, p.proj_g2, p.proj_be2, sMean, sRstd, lane);
}

extern "C" void kernel_launch(void* const* d_in, const int* in_sizes, int n_in,
                              void* d_out, int out_size, void* d_ws, size_t ws_size,
                              hipStream_t stream) {
    const int N = in_sizes[0];          // 131072, divisible by 16

    // Workspace layout (f16 elements): transposed f16 weights, ~2.0 MB total.
    _Float16* ws       = (_Float16*)d_ws;
    _Float16* wt_fus1  = ws;                       // 256*160  = 40960
    _Float16* wt_fus2  = ws + 40960;               // 768*256  = 196608
    _Float16* wt_proj1 = ws + 40960 + 196608;      // 512*768  = 393216
    _Float16* wt_proj2 = ws + 40960 + 196608 + 393216;  // 768*512

    auto launchT = [&](const void* src, _Float16* dst, int K, int Nout) {
        int total = K * Nout;
        transpose_f32_to_f16<<<(total + 255) / 256, 256, 0, stream>>>(
            (const float*)src, dst, K, Nout);
    };
    launchT(d_in[12], wt_fus1,  160, 256);
    launchT(d_in[16], wt_fus2,  256, 768);
    launchT(d_in[20], wt_proj1, 768, 512);
    launchT(d_in[24], wt_proj2, 512, 768);

    Params p;
    p.comp_ids     = (const int*)  d_in[0];
    p.change_ids   = (const int*)  d_in[1];
    p.bbox         = (const float*)d_in[2];
    p.comp_table   = (const float*)d_in[4];
    p.change_table = (const float*)d_in[5];
    p.pos_w1 = (const float*)d_in[6];   p.pos_b1 = (const float*)d_in[7];
    p.pos_g1 = (const float*)d_in[8];   p.pos_be1 = (const float*)d_in[9];
    p.pos_w2 = (const float*)d_in[10];  p.pos_b2 = (const float*)d_in[11];
    p.fus_b1  = (const float*)d_in[13]; p.fus_g1  = (const float*)d_in[14]; p.fus_be1 = (const float*)d_in[15];
    p.fus_b2  = (const float*)d_in[17]; p.fus_g2  = (const float*)d_in[18]; p.fus_be2 = (const float*)d_in[19];
    p.proj_b1 = (const float*)d_in[21]; p.proj_g1 = (const float*)d_in[22]; p.proj_be1 = (const float*)d_in[23];
    p.proj_b2 = (const float*)d_in[25]; p.proj_g2 = (const float*)d_in[26]; p.proj_be2 = (const float*)d_in[27];
    p.wt_fus1 = wt_fus1; p.wt_fus2 = wt_fus2;
    p.wt_proj1 = wt_proj1; p.wt_proj2 = wt_proj2;
    p.out = (float*)d_out;

    phrase_encoder_kernel<<<N / 16, 32, 0, stream>>>(p);

    // Second tuple output: batch_ids passthrough (int32 bits) appended after y.
    hipMemcpyAsync((float*)d_out + (size_t)N * 768, d_in[3],
                   (size_t)N * sizeof(int), hipMemcpyDeviceToDevice, stream);
}
